// DecoderLayer_72026601554425
// MI455X (gfx1250) — matrix operand order repaired and checked by reading, not compiled, not run on hardware
//
#include <hip/hip_runtime.h>
#include <cstdint>
#include <cstddef>

// ---------------------------------------------------------------------------
// Decoder layer (B=4, SQ=SK=1024, D=1024, H=16, DFF=4096) for MI455X/gfx1250.
// bf16 WMMA (v_wmma_f32_16x16x32_bf16) for every GEMM, fp32 LN/softmax/residual.
// GEMM uses a double-buffered GLOBAL_LOAD_ASYNC_TO_LDS pipeline (ASYNCcnt).
// ---------------------------------------------------------------------------

#define DEV_ __device__ __forceinline__

typedef __attribute__((ext_vector_type(16))) __bf16 bf16x16;
typedef __attribute__((ext_vector_type(8)))  float  f32x8;
typedef int i32x4v __attribute__((vector_size(16)));  // matches builtin param type

union FragU { uint4 u[2]; bf16x16 v; };

#if defined(__AMDGCN__) && __has_builtin(__builtin_amdgcn_global_load_async_to_lds_b128)
#define HAVE_ASYNC_LDS 1
#else
#define HAVE_ASYNC_LDS 0
#endif

// 16-byte global -> LDS copy. Async path: DMA into LDS tracked by ASYNCcnt,
// no VGPR round trip. Fallback: plain load+store.
DEV_ void copy16_g2l(uint16_t* lds_dst, const uint16_t* gsrc) {
#if HAVE_ASYNC_LDS
  __builtin_amdgcn_global_load_async_to_lds_b128(
      (__attribute__((address_space(1))) i32x4v*)gsrc,
      (__attribute__((address_space(3))) i32x4v*)lds_dst, 0, 0);
#else
  *(uint4*)lds_dst = *(const uint4*)gsrc;
#endif
}

DEV_ void wait_async_all() {
#if HAVE_ASYNC_LDS
#if __has_builtin(__builtin_amdgcn_s_wait_asynccnt)
  __builtin_amdgcn_s_wait_asynccnt(0);
#else
  asm volatile("s_wait_asynccnt 0x0" ::: "memory");
#endif
#endif
}

DEV_ uint16_t f2bf(float f) {
  union { float f; uint32_t u; } c; c.f = f;
  uint32_t u = c.u;
  uint32_t r = (u + 0x7FFFu + ((u >> 16) & 1u)) >> 16;  // round-to-nearest-even
  return (uint16_t)r;
}

// A fragment, 16x32 bf16 (MxK): lane row = lr, elems 0..7 -> K=sel*8..+7,
// elems 8..15 -> K=16+sel*8..+7   (ISA 7.12.2, 16-bit A 16x32 layout)
DEV_ bf16x16 frag_a(const uint16_t* rowp, int sel) {
  FragU f;
  f.u[0] = *(const uint4*)(rowp + sel * 8);
  f.u[1] = *(const uint4*)(rowp + sel * 8 + 16);
  return f.v;
}

// B fragment, 32x16 bf16 (KxN), source stored as Bt[N][K]: lane col = lr,
// elems 0..7 -> K = sel*8..+7, elems 8..15 -> K = 16+sel*8..+7
DEV_ bf16x16 frag_b(const uint16_t* rowp, int sel) {
  FragU f;
  f.u[0] = *(const uint4*)(rowp + sel * 8);
  f.u[1] = *(const uint4*)(rowp + sel * 8 + 16);
  return f.v;
}

// bf16 output layout modes
#define BMODE_PLAIN       0  // [M,N]
#define BMODE_HEADSPLIT   1  // [B,H,S,64]   (q/k)
#define BMODE_HEADSPLIT_T 2  // [B,H,64,S]   (v transposed)
#define BMODE_MERGE       3  // batched z=(b,h): merged[(b*1024+s)*1024 + h*64+n]

// ---------------------------------------------------------------------------
// NT WMMA GEMM: C[M,N] = A[M,K] * Bt[N,K]^T, bf16 in, fp32 accumulate.
// BM=128, BK=32, BN template (128/64). 256 threads = 8 waves (wave32),
// wave grid 2(M)x4(N), 4 x NT accumulator tiles of 16x16 per wave.
// Double-buffered async-to-LDS staging: issue tile k+1, compute tile k,
// s_wait_asynccnt 0 + barrier, swap.
// ---------------------------------------------------------------------------
template <int BN>
__global__ __launch_bounds__(256) void wmma_gemm_nt(
    const uint16_t* __restrict__ A, const uint16_t* __restrict__ Bt,
    int M, int N, int K,
    long long sA, long long sB, long long sC,
    const float* __restrict__ bias, const float* __restrict__ resid,
    float* __restrict__ outf, uint16_t* __restrict__ outb, int bmode,
    int relu, float scale, const float* __restrict__ mask, int mask_mode) {
  constexpr int BM = 128, BK = 32;
  constexpr int SPAN = BN / 4;        // N span per wave
  constexpr int NT = SPAN / 16;       // 16x16 n-tiles per wave
  constexpr int LDT = BK + 8;         // padded LDS row stride (80B, 16B aligned)
  constexpr int CA = (BM * 4) / 256;  // A chunks per thread
  constexpr int CB = (BN * 4) / 256;  // B chunks per thread
  __shared__ uint16_t sa[2][BM * LDT];
  __shared__ uint16_t sb[2][BN * LDT];

  const int z = blockIdx.z;
  A += (long long)z * sA;
  Bt += (long long)z * sB;

  const int m0 = blockIdx.y * BM;
  const int n0 = blockIdx.x * BN;
  const int t = threadIdx.x;
  const int w = t >> 5, lane = t & 31;
  const int lr = lane & 15, sel = lane >> 4;
  const int wm = w & 1, wn = w >> 1;

  // Per-thread staging addresses (computed once, bumped by BK per k-step).
  const uint16_t* ga[CA]; int la[CA];
  const uint16_t* gb[CB]; int lb[CB];
#pragma unroll
  for (int i = 0; i < CA; i++) {
    int c = t + i * 256, r = c >> 2, kc = (c & 3) * 8;
    ga[i] = A + (long long)(m0 + r) * K + kc;
    la[i] = r * LDT + kc;
  }
#pragma unroll
  for (int i = 0; i < CB; i++) {
    int c = t + i * 256, r = c >> 2, kc = (c & 3) * 8;
    gb[i] = Bt + (long long)(n0 + r) * K + kc;
    lb[i] = r * LDT + kc;
  }

  auto issue_tiles = [&](int buf) {
#pragma unroll
    for (int i = 0; i < CA; i++) { copy16_g2l(&sa[buf][la[i]], ga[i]); ga[i] += BK; }
#pragma unroll
    for (int i = 0; i < CB; i++) { copy16_g2l(&sb[buf][lb[i]], gb[i]); gb[i] += BK; }
  };

  issue_tiles(0);
  wait_async_all();
  __syncthreads();

  f32x8 acc[4][NT] = {};
  int buf = 0;
  for (int kb = 0; kb < K; kb += BK) {
    if (kb + BK < K) issue_tiles(buf ^ 1);  // overlap copy of next tile

    bf16x16 bfr[NT];
#pragma unroll
    for (int nt = 0; nt < NT; nt++)
      bfr[nt] = frag_b(&sb[buf][(wn * SPAN + nt * 16 + lr) * LDT], sel);
#pragma unroll
    for (int mt = 0; mt < 4; mt++) {
      bf16x16 af = frag_a(&sa[buf][(wm * 64 + mt * 16 + lr) * LDT], sel);
#pragma unroll
      for (int nt = 0; nt < NT; nt++)
        acc[mt][nt] = __builtin_amdgcn_wmma_f32_16x16x32_bf16(
            false, af, false, bfr[nt], (short)0, acc[mt][nt], false, false);
    }

    wait_async_all();  // my async copies into buf^1 have landed
    __syncthreads();   // everyone done reading buf & done copying
    buf ^= 1;
  }

  // Epilogue. C 16x16 layout: lane col = lr, vgpr i -> row = i + sel*8.
#pragma unroll
  for (int mt = 0; mt < 4; mt++) {
#pragma unroll
    for (int nt = 0; nt < NT; nt++) {
#pragma unroll
      for (int i = 0; i < 8; i++) {
        int row = m0 + wm * 64 + mt * 16 + sel * 8 + i;
        int col = n0 + wn * SPAN + nt * 16 + lr;
        float v = acc[mt][nt][i];
        if (bias) v += bias[col];
        if (scale != 1.0f) v *= scale;
        if (mask_mode == 1) v += mask[col] * -1e9f;                            // [1,1,1,SK]
        else if (mask_mode == 2) v += mask[(long long)row * N + col] * -1e9f;  // [SQ,SQ]
        if (relu) v = v > 0.f ? v : 0.f;
        if (resid) v += resid[(long long)row * N + col];
        if (outf) outf[(long long)z * sC + (long long)row * N + col] = v;
        if (outb) {
          long long di;
          if (bmode == BMODE_PLAIN) {
            di = (long long)row * N + col;
          } else if (bmode == BMODE_HEADSPLIT) {
            int b = row >> 10, s = row & 1023, h = col >> 6, dep = col & 63;
            di = (((long long)(b * 16 + h) * 1024 + s) * 64) + dep;
          } else if (bmode == BMODE_HEADSPLIT_T) {
            int b = row >> 10, s = row & 1023, h = col >> 6, dep = col & 63;
            di = (((long long)(b * 16 + h) * 64 + dep) * 1024) + s;
          } else {  // BMODE_MERGE: z = b*16+h, row = s, col = dep
            int b = z >> 4, h = z & 15;
            di = ((long long)(b * 1024 + row) * 1024) + h * 64 + col;
          }
          outb[di] = f2bf(v);
        }
      }
    }
  }
}

// ---------------------------------------------------------------------------
// Transpose-cast: fp32 [K,N] -> bf16 [N,K]
// ---------------------------------------------------------------------------
__global__ __launch_bounds__(256) void transpose_cast_kernel(
    const float* __restrict__ in, uint16_t* __restrict__ out, int K, int N) {
  __shared__ float tile[32][33];
  const int nb = blockIdx.x * 32, kb = blockIdx.y * 32;
  const int tx = threadIdx.x & 31, ty = threadIdx.x >> 5;
#pragma unroll
  for (int i = 0; i < 4; i++)
    tile[ty + i * 8][tx] = in[(long long)(kb + ty + i * 8) * N + nb + tx];
  __syncthreads();
#pragma unroll
  for (int i = 0; i < 4; i++)
    out[(long long)(nb + ty + i * 8) * K + kb + tx] = f2bf(tile[tx][ty + i * 8]);
}

__global__ __launch_bounds__(256) void cast_bf16_kernel(
    const float* __restrict__ in, uint16_t* __restrict__ out, long long n4) {
  long long i = ((long long)blockIdx.x * 256 + threadIdx.x) * 4;
  if (i < n4) {
    float4 v = *(const float4*)(in + i);
    *(ushort4*)(out + i) = make_ushort4(f2bf(v.x), f2bf(v.y), f2bf(v.z), f2bf(v.w));
  }
}

// ---------------------------------------------------------------------------
// LayerNorm over D=1024, fp32 in -> bf16 out. One block (256 thr, 8 waves) per row.
// ---------------------------------------------------------------------------
__global__ __launch_bounds__(256) void layernorm_bf16_kernel(
    const float* __restrict__ x, const float* __restrict__ g,
    const float* __restrict__ b, uint16_t* __restrict__ out) {
  const long long row = blockIdx.x;
  const int t = threadIdx.x;
  const float4 v = *(const float4*)(x + row * 1024 + t * 4);
  float s = v.x + v.y + v.z + v.w;
  float ss = v.x * v.x + v.y * v.y + v.z * v.z + v.w * v.w;
#pragma unroll
  for (int o = 16; o > 0; o >>= 1) {  // wave32 reduce
    s += __shfl_xor(s, o, 32);
    ss += __shfl_xor(ss, o, 32);
  }
  __shared__ float rs[8], rss[8];
  const int w = t >> 5, lane = t & 31;
  if (lane == 0) { rs[w] = s; rss[w] = ss; }
  __syncthreads();
  if (w == 0) {
    float a = (lane < 8) ? rs[lane] : 0.f;
    float a2 = (lane < 8) ? rss[lane] : 0.f;
#pragma unroll
    for (int o = 4; o > 0; o >>= 1) { a += __shfl_xor(a, o, 32); a2 += __shfl_xor(a2, o, 32); }
    if (lane == 0) { rs[0] = a; rss[0] = a2; }
  }
  __syncthreads();
  const float mu = rs[0] * (1.f / 1024.f);
  const float var = rss[0] * (1.f / 1024.f) - mu * mu;
  const float rinv = rsqrtf(var + 1e-6f);
  const float4 gv = *(const float4*)(g + t * 4);
  const float4 bv = *(const float4*)(b + t * 4);
  uint16_t* op = out + row * 1024 + t * 4;
  *(ushort4*)op = make_ushort4(f2bf((v.x - mu) * rinv * gv.x + bv.x),
                               f2bf((v.y - mu) * rinv * gv.y + bv.y),
                               f2bf((v.z - mu) * rinv * gv.z + bv.z),
                               f2bf((v.w - mu) * rinv * gv.w + bv.w));
}

// ---------------------------------------------------------------------------
// Softmax over last dim (1024). In-place fp32 (d_out attn slice) + bf16 copy.
// ---------------------------------------------------------------------------
__global__ __launch_bounds__(256) void softmax_kernel(
    float* __restrict__ p, uint16_t* __restrict__ pb) {
  const long long row = blockIdx.x;
  float* rp = p + row * 1024;
  const int t = threadIdx.x;
  float4 v = *(const float4*)(rp + t * 4);
  float mx = fmaxf(fmaxf(v.x, v.y), fmaxf(v.z, v.w));
#pragma unroll
  for (int o = 16; o > 0; o >>= 1) mx = fmaxf(mx, __shfl_xor(mx, o, 32));
  __shared__ float r[8];
  const int w = t >> 5, lane = t & 31;
  if (lane == 0) r[w] = mx;
  __syncthreads();
  if (w == 0) {
    float a = (lane < 8) ? r[lane] : -3.4e38f;
#pragma unroll
    for (int o = 4; o > 0; o >>= 1) a = fmaxf(a, __shfl_xor(a, o, 32));
    if (lane == 0) r[0] = a;
  }
  __syncthreads();
  mx = r[0];
  __syncthreads();
  v.x = __expf(v.x - mx); v.y = __expf(v.y - mx);
  v.z = __expf(v.z - mx); v.w = __expf(v.w - mx);
  float s = v.x + v.y + v.z + v.w;
#pragma unroll
  for (int o = 16; o > 0; o >>= 1) s += __shfl_xor(s, o, 32);
  if (lane == 0) r[w] = s;
  __syncthreads();
  if (w == 0) {
    float a = (lane < 8) ? r[lane] : 0.f;
#pragma unroll
    for (int o = 4; o > 0; o >>= 1) a += __shfl_xor(a, o, 32);
    if (lane == 0) r[0] = a;
  }
  __syncthreads();
  const float inv = 1.f / r[0];
  v.x *= inv; v.y *= inv; v.z *= inv; v.w *= inv;
  *(float4*)(rp + t * 4) = v;
  *(ushort4*)(pb + row * 1024 + t * 4) =
      make_ushort4(f2bf(v.x), f2bf(v.y), f2bf(v.z), f2bf(v.w));
}

// ---------------------------------------------------------------------------
// Host-side orchestration
// ---------------------------------------------------------------------------
extern "C" void kernel_launch(void* const* d_in, const int* in_sizes, int n_in,
                              void* d_out, int out_size, void* d_ws, size_t ws_size,
                              hipStream_t stream) {
  (void)in_sizes; (void)n_in; (void)out_size; (void)ws_size;
  constexpr int Bz = 4, SQ = 1024, D = 1024, H = 16, DFF = 4096;
  constexpr long long MD = (long long)Bz * SQ;  // 4096 rows

  const float* x     = (const float*)d_in[0];
  const float* y     = (const float*)d_in[1];
  const float* cmask = (const float*)d_in[2];   // [1,1,1,SK]
  const float* pmask = (const float*)d_in[3];   // [1,1,SQ,SQ]
  const float* W[10] = { (const float*)d_in[4],  (const float*)d_in[6],
                         (const float*)d_in[8],  (const float*)d_in[10],
                         (const float*)d_in[12], (const float*)d_in[14],
                         (const float*)d_in[16], (const float*)d_in[18],
                         (const float*)d_in[20], (const float*)d_in[22] };
  const float* Bias[10] = { (const float*)d_in[5],  (const float*)d_in[7],
                            (const float*)d_in[9],  (const float*)d_in[11],
                            (const float*)d_in[13], (const float*)d_in[15],
                            (const float*)d_in[17], (const float*)d_in[19],
                            (const float*)d_in[21], (const float*)d_in[23] };
  const float* ln_g[3] = { (const float*)d_in[24], (const float*)d_in[26], (const float*)d_in[28] };
  const float* ln_b[3] = { (const float*)d_in[25], (const float*)d_in[27], (const float*)d_in[29] };

  float* out_final = (float*)d_out;
  float* dec_attn  = out_final + MD * D;                       // [B,H,SQ,SK] fp32
  float* dec_eattn = dec_attn + (long long)Bz * H * SQ * SQ;   // [B,H,SQ,SK] fp32

  // workspace carve (256B aligned)
  char* wsp = (char*)d_ws;
  size_t off = 0;
  auto carve = [&](size_t bytes) -> void* {
    void* p = wsp + off;
    off += (bytes + 255) & ~(size_t)255;
    return p;
  };
  uint16_t* wt[10];
  for (int i = 0; i < 8; i++) wt[i] = (uint16_t*)carve((size_t)D * D * 2);
  wt[8] = (uint16_t*)carve((size_t)DFF * D * 2);  // w1t [DFF, D]
  wt[9] = (uint16_t*)carve((size_t)D * DFF * 2);  // w2t [D, DFF]
  uint16_t* y_bf      = (uint16_t*)carve((size_t)MD * D * 2);
  uint16_t* ln_bf     = (uint16_t*)carve((size_t)MD * D * 2);
  uint16_t* q_bf      = (uint16_t*)carve((size_t)MD * D * 2);
  uint16_t* k_bf      = (uint16_t*)carve((size_t)MD * D * 2);
  uint16_t* vT_bf     = (uint16_t*)carve((size_t)MD * D * 2);
  uint16_t* probs_bf  = (uint16_t*)carve((size_t)Bz * H * SQ * SQ * 2);
  uint16_t* merged_bf = (uint16_t*)carve((size_t)MD * D * 2);
  uint16_t* h1_bf     = (uint16_t*)carve((size_t)MD * DFF * 2);
  float*    r1        = (float*)carve((size_t)MD * D * 4);
  float*    r2        = (float*)carve((size_t)MD * D * 4);

  auto gemm = [&](int bn, const uint16_t* A, const uint16_t* Bt, int M, int N,
                  int K, int batch, long long sA, long long sB, long long sC,
                  const float* bias, const float* resid, float* outf,
                  uint16_t* outb, int bmode, int relu, float scale,
                  const float* mask, int mmode) {
    if (bn == 128) {
      dim3 g(N / 128, M / 128, batch);
      wmma_gemm_nt<128><<<g, 256, 0, stream>>>(A, Bt, M, N, K, sA, sB, sC, bias,
                                               resid, outf, outb, bmode, relu,
                                               scale, mask, mmode);
    } else {
      dim3 g(N / 64, M / 128, batch);
      wmma_gemm_nt<64><<<g, 256, 0, stream>>>(A, Bt, M, N, K, sA, sB, sC, bias,
                                              resid, outf, outb, bmode, relu,
                                              scale, mask, mmode);
    }
  };

  // 1) weights -> bf16 [N,K]
  for (int i = 0; i < 8; i++)
    transpose_cast_kernel<<<dim3(D / 32, D / 32), 256, 0, stream>>>(W[i], wt[i], D, D);
  transpose_cast_kernel<<<dim3(DFF / 32, D / 32), 256, 0, stream>>>(W[8], wt[8], D, DFF);
  transpose_cast_kernel<<<dim3(D / 32, DFF / 32), 256, 0, stream>>>(W[9], wt[9], DFF, D);
  cast_bf16_kernel<<<(MD * D) / 1024, 256, 0, stream>>>(y, y_bf, MD * D);

  const float isq = 0.125f;  // 1/sqrt(64)

  // ---- self-attention sub-layer (mask = padding_mask, 2D) ----
  layernorm_bf16_kernel<<<MD, 256, 0, stream>>>(x, ln_g[0], ln_b[0], ln_bf);
  gemm(128, ln_bf, wt[0], MD, D, D, 1, 0, 0, 0, Bias[0], nullptr, nullptr, q_bf,  BMODE_HEADSPLIT,   0, 1.f, nullptr, 0);
  gemm(128, ln_bf, wt[1], MD, D, D, 1, 0, 0, 0, Bias[1], nullptr, nullptr, k_bf,  BMODE_HEADSPLIT,   0, 1.f, nullptr, 0);
  gemm(128, ln_bf, wt[2], MD, D, D, 1, 0, 0, 0, Bias[2], nullptr, nullptr, vT_bf, BMODE_HEADSPLIT_T, 0, 1.f, nullptr, 0);
  gemm(128, q_bf, k_bf, SQ, SQ, 64, Bz * H, (long long)SQ * 64, (long long)SQ * 64,
       (long long)SQ * SQ, nullptr, nullptr, dec_attn, nullptr, 0, 0, isq, pmask, 2);
  softmax_kernel<<<(long long)Bz * H * SQ, 256, 0, stream>>>(dec_attn, probs_bf);
  gemm(64, probs_bf, vT_bf, SQ, 64, SQ, Bz * H, (long long)SQ * SQ, (long long)64 * SQ,
       0, nullptr, nullptr, nullptr, merged_bf, BMODE_MERGE, 0, 1.f, nullptr, 0);
  gemm(128, merged_bf, wt[3], MD, D, D, 1, 0, 0, 0, Bias[3], x, r1, nullptr, 0, 0, 1.f, nullptr, 0);

  // ---- cross-attention sub-layer (K/V from y; mask = causality, 1D) ----
  layernorm_bf16_kernel<<<MD, 256, 0, stream>>>(r1, ln_g[1], ln_b[1], ln_bf);
  gemm(128, ln_bf, wt[4], MD, D, D, 1, 0, 0, 0, Bias[4], nullptr, nullptr, q_bf,  BMODE_HEADSPLIT,   0, 1.f, nullptr, 0);
  gemm(128, y_bf,  wt[5], MD, D, D, 1, 0, 0, 0, Bias[5], nullptr, nullptr, k_bf,  BMODE_HEADSPLIT,   0, 1.f, nullptr, 0);
  gemm(128, y_bf,  wt[6], MD, D, D, 1, 0, 0, 0, Bias[6], nullptr, nullptr, vT_bf, BMODE_HEADSPLIT_T, 0, 1.f, nullptr, 0);
  gemm(128, q_bf, k_bf, SQ, SQ, 64, Bz * H, (long long)SQ * 64, (long long)SQ * 64,
       (long long)SQ * SQ, nullptr, nullptr, dec_eattn, nullptr, 0, 0, isq, cmask, 1);
  softmax_kernel<<<(long long)Bz * H * SQ, 256, 0, stream>>>(dec_eattn, probs_bf);
  gemm(64, probs_bf, vT_bf, SQ, 64, SQ, Bz * H, (long long)SQ * SQ, (long long)64 * SQ,
       0, nullptr, nullptr, nullptr, merged_bf, BMODE_MERGE, 0, 1.f, nullptr, 0);
  gemm(128, merged_bf, wt[7], MD, D, D, 1, 0, 0, 0, Bias[7], r1, r2, nullptr, 0, 0, 1.f, nullptr, 0);

  // ---- FFN sub-layer ----
  layernorm_bf16_kernel<<<MD, 256, 0, stream>>>(r2, ln_g[2], ln_b[2], ln_bf);
  gemm(128, ln_bf, wt[8], MD, DFF, D, 1, 0, 0, 0, Bias[8], nullptr, nullptr, h1_bf,
       BMODE_PLAIN, /*relu=*/1, 1.f, nullptr, 0);
  gemm(128, h1_bf, wt[9], MD, D, DFF, 1, 0, 0, 0, Bias[9], r2, out_final, nullptr,
       0, 0, 1.f, nullptr, 0);
}
